// LapNetWaveFunction_72971494359227
// MI455X (gfx1250) — matrix operand order repaired and checked by reading, not compiled, not run on hardware
//
#include <hip/hip_runtime.h>
#include <hip/hip_bf16.h>
#include <cstdint>

// LapNet wave function forward for MI455X (gfx1250, wave32, WMMA + async LDS).
// B=2048, NE=32 (NU=ND=16), A=4, D=256, L=4, H=8, DH=32, K=16 dets.

typedef _Float16 f16;
typedef __attribute__((ext_vector_type(16))) _Float16 v16h;
typedef __attribute__((ext_vector_type(8)))  _Float16 v8h;
typedef __attribute__((ext_vector_type(8)))  float    v8f;

constexpr int BATCH = 2048;
constexpr int NEL   = 32;
constexpr int NUP   = 16;
constexpr int NATM  = 4;
constexpr int DMODEL= 256;
constexpr int NLAY  = 4;
constexpr int NDET  = 16;
constexpr int ROWS  = BATCH * NEL;     // 65536
constexpr int HROWS = BATCH * NUP;     // 32768

// ---------------------------------------------------------------------------
// CDNA5 async global->LDS copy (16 B per lane), tracked by ASYNCcnt.
// LDS address = low 32 bits of the shared-aperture generic pointer.
// ---------------------------------------------------------------------------
__device__ __forceinline__ void async_cp16(f16* ldsDst, const f16* g) {
  uint32_t l = (uint32_t)(uintptr_t)ldsDst;
  asm volatile("global_load_async_to_lds_b128 %0, %1, off"
               :: "v"(l), "v"(g) : "memory");
}

// ---------------------------------------------------------------------------
// Weight convert: f32 [Din][Dout] -> f16 transposed [Dout][Din] (B-fragment
// friendly: per-output-column K is contiguous).
// ---------------------------------------------------------------------------
__global__ void conv_t_kernel(const float* __restrict__ src, f16* __restrict__ dst,
                              int Din, int Dout) {
  int t = blockIdx.x * 256 + threadIdx.x;
  if (t >= Din * Dout) return;
  int o = t / Din, i = t - o * Din;
  dst[t] = (f16)src[(size_t)i * Dout + o];
}

// ---------------------------------------------------------------------------
// Features: ae, r_ae, spin -> h0 = feat @ W_in + b_in ; env = sum pi*exp(-sig*r)
// ---------------------------------------------------------------------------
__global__ void features_kernel(const float* __restrict__ ec, const float* __restrict__ spins,
                                const float* __restrict__ atoms, const float* __restrict__ Win,
                                const float* __restrict__ bin, const float* __restrict__ sigma,
                                const float* __restrict__ pi_env,
                                float* __restrict__ h, float* __restrict__ env) {
  int idx = blockIdx.x * 256 + threadIdx.x;
  if (idx >= ROWS) return;
  float ex = ec[idx * 3 + 0], ey = ec[idx * 3 + 1], ez = ec[idx * 3 + 2];
  float feat[17];
  float envv = 0.f;
  #pragma unroll
  for (int a = 0; a < NATM; ++a) {
    float dx = ex - atoms[a * 3 + 0];
    float dy = ey - atoms[a * 3 + 1];
    float dz = ez - atoms[a * 3 + 2];
    feat[a * 3 + 0] = dx; feat[a * 3 + 1] = dy; feat[a * 3 + 2] = dz;
    float r = sqrtf(dx * dx + dy * dy + dz * dz + 1e-12f);
    feat[12 + a] = r;
    envv += pi_env[a] * __expf(-sigma[a] * r);
  }
  feat[16] = spins[idx];
  env[idx] = envv;
  float* hp = h + (size_t)idx * DMODEL;
  for (int d = 0; d < DMODEL; ++d) {
    float acc = bin[d];
    #pragma unroll
    for (int i = 0; i < 17; ++i) acc += feat[i] * Win[i * DMODEL + d];
    hp[d] = acc;
  }
}

__global__ void f32_to_f16_kernel(const float* __restrict__ x, f16* __restrict__ y, size_t n) {
  size_t i = (size_t)blockIdx.x * 256 + threadIdx.x;
  size_t stride = (size_t)gridDim.x * 256;
  for (; i < n; i += stride) y[i] = (f16)x[i];
}

// ---------------------------------------------------------------------------
// WMMA GEMM: C[M x N] = epi(A[M x K] * Bt[N x K]^T + bias).
// Block tile 128x64, 8 waves, each wave 16x64 (4 accumulators), k-step 32.
// Double-buffered LDS tiles filled with GLOBAL_LOAD_ASYNC_TO_LDS_B128;
// tile k+1 streams in (3 async ops/wave) while tile k runs 4 WMMAs/wave.
// ---------------------------------------------------------------------------
constexpr int EPI_F16 = 0, EPI_TANH_F16 = 1, EPI_TANH_ADD = 2, EPI_GELU_F16 = 3;

template<int EPI>
__global__ __launch_bounds__(256) void gemm_wmma_kernel(
    const f16* __restrict__ A, const f16* __restrict__ Bt,
    const float* __restrict__ bias,
    f16* __restrict__ Cf16, float* __restrict__ Cacc,
    int M, int N, int K) {
  __shared__ __align__(16) f16 As[2][128 * 48];   // rows padded to 96 B
  __shared__ __align__(16) f16 Bs[2][64 * 48];
  const int tid  = threadIdx.x;
  const int lane = tid & 31, wv = tid >> 5;
  const bool hi  = lane >= 16;
  const int l16  = lane & 15;
  const int n0   = blockIdx.x * 64;
  const int m0   = blockIdx.y * 128;

  // static per-thread chunk assignment (16 B chunks)
  const int arow0 = tid >> 2,          ach0 = tid & 3;          // A chunk 1
  const int arow1 = (tid + 256) >> 2,  ach1 = tid & 3;          // A chunk 2
  const int brow  = tid >> 2,          bch  = tid & 3;          // B chunk

  v8f acc[4];
  const v8f zero = {0.f, 0.f, 0.f, 0.f, 0.f, 0.f, 0.f, 0.f};
  #pragma unroll
  for (int nt = 0; nt < 4; ++nt) acc[nt] = zero;

  const int ksteps = K >> 5;

  // prime buffer 0
  async_cp16(&As[0][arow0 * 48 + ach0 * 8], A + (size_t)(m0 + arow0) * K + ach0 * 8);
  async_cp16(&As[0][arow1 * 48 + ach1 * 8], A + (size_t)(m0 + arow1) * K + ach1 * 8);
  async_cp16(&Bs[0][brow  * 48 + bch  * 8], Bt + (size_t)(n0 + brow) * K + bch * 8);

  for (int ks = 0; ks < ksteps; ++ks) {
    const int cb = ks & 1;
    if (ks + 1 < ksteps) {
      const int k1 = (ks + 1) * 32;
      async_cp16(&As[cb ^ 1][arow0 * 48 + ach0 * 8], A + (size_t)(m0 + arow0) * K + k1 + ach0 * 8);
      async_cp16(&As[cb ^ 1][arow1 * 48 + ach1 * 8], A + (size_t)(m0 + arow1) * K + k1 + ach1 * 8);
      async_cp16(&Bs[cb ^ 1][brow  * 48 + bch  * 8], Bt + (size_t)(n0 + brow) * K + k1 + bch * 8);
      asm volatile("s_wait_asynccnt 0x3" ::: "memory");   // tile ks landed
    } else {
      asm volatile("s_wait_asynccnt 0x0" ::: "memory");
    }
    __syncthreads();

    union { v16h v; v8h h[2]; } fa;
    const f16* ap = As[cb] + (wv * 16 + l16) * 48 + (hi ? 8 : 0);
    fa.h[0] = *(const v8h*)(ap);        // K = base .. base+7
    fa.h[1] = *(const v8h*)(ap + 16);   // K = base+16 .. base+23
    #pragma unroll
    for (int nt = 0; nt < 4; ++nt) {
      union { v16h v; v8h h[2]; } fb;
      const f16* bp = Bs[cb] + (nt * 16 + l16) * 48 + (hi ? 8 : 0);
      fb.h[0] = *(const v8h*)(bp);
      fb.h[1] = *(const v8h*)(bp + 16);
      acc[nt] = __builtin_amdgcn_wmma_f32_16x16x32_f16(
          false, fa.v, false, fb.v, (short)0, acc[nt], false, false);
    }
    __syncthreads();   // all waves done reading cb before it is refilled
  }

  #pragma unroll
  for (int nt = 0; nt < 4; ++nt) {
    int n = n0 + nt * 16 + l16;
    float bv = bias ? bias[n] : 0.f;
    #pragma unroll
    for (int r = 0; r < 8; ++r) {
      int m = m0 + wv * 16 + r + (hi ? 8 : 0);
      float x = acc[nt][r] + bv;
      size_t idx = (size_t)m * N + n;
      if constexpr (EPI == EPI_F16)       Cf16[idx] = (f16)x;
      else if constexpr (EPI == EPI_TANH_F16) Cf16[idx] = (f16)tanhf(x);
      else if constexpr (EPI == EPI_TANH_ADD) Cacc[idx] += tanhf(x);
      else if constexpr (EPI == EPI_GELU_F16)
        Cf16[idx] = (f16)(0.5f * x * (1.f + erff(x * 0.70710678118654752f)));
    }
  }
}

// ---------------------------------------------------------------------------
// Attention: one block per batch element, one wave per head, one lane per
// electron (NE == 32 == wave32). qkv packed [row][768] = q|k|v.
// ---------------------------------------------------------------------------
__global__ __launch_bounds__(256) void attn_kernel(const f16* __restrict__ qkv,
                                                   f16* __restrict__ o) {
  int b = blockIdx.x;
  int tid = threadIdx.x, wv = tid >> 5, lane = tid & 31;
  __shared__ __align__(16) f16 ks[8][32][32];
  __shared__ __align__(16) f16 vs[8][32][32];
  const size_t rs = 768;
  const f16* kp = qkv + ((size_t)b * NEL + lane) * rs + 256 + wv * 32;
  const f16* vp = kp + 256;
  #pragma unroll
  for (int c = 0; c < 4; ++c) {
    *(v8h*)&ks[wv][lane][c * 8] = *(const v8h*)(kp + c * 8);
    *(v8h*)&vs[wv][lane][c * 8] = *(const v8h*)(vp + c * 8);
  }
  float q[32];
  const f16* qp = qkv + ((size_t)b * NEL + lane) * rs + wv * 32;
  #pragma unroll
  for (int d = 0; d < 32; ++d) q[d] = (float)qp[d];
  __syncthreads();

  float lg[32];
  float mx = -3.0e38f;
  for (int m = 0; m < 32; ++m) {
    float s = 0.f;
    #pragma unroll
    for (int d = 0; d < 32; ++d) s += q[d] * (float)ks[wv][m][d];
    s *= 0.17677669529663689f;   // 1/sqrt(DH=32)
    lg[m] = s;
    mx = fmaxf(mx, s);
  }
  float sum = 0.f;
  for (int m = 0; m < 32; ++m) { lg[m] = __expf(lg[m] - mx); sum += lg[m]; }
  float inv = 1.f / sum;
  float oo[32];
  #pragma unroll
  for (int d = 0; d < 32; ++d) oo[d] = 0.f;
  for (int m = 0; m < 32; ++m) {
    float p = lg[m] * inv;
    #pragma unroll
    for (int d = 0; d < 32; ++d) oo[d] += p * (float)vs[wv][m][d];
  }
  f16* op = o + ((size_t)b * NEL + lane) * DMODEL + wv * 32;
  #pragma unroll
  for (int d = 0; d < 32; ++d) op[d] = (f16)oo[d];
}

// ---------------------------------------------------------------------------
// Split final h into compact up/down f16 blocks.
// ---------------------------------------------------------------------------
__global__ void split_kernel(const float* __restrict__ h, f16* __restrict__ hfu,
                             f16* __restrict__ hfd) {
  size_t idx = (size_t)blockIdx.x * 256 + threadIdx.x;
  if (idx >= (size_t)HROWS * DMODEL) return;
  int d = (int)(idx & 255);
  int m = (int)(idx >> 8);
  int b = m >> 4, e = m & 15;
  hfu[idx] = (f16)h[((size_t)(b * NEL + e)) * DMODEL + d];
  hfd[idx] = (f16)h[((size_t)(b * NEL + NUP + e)) * DMODEL + d];
}

// ---------------------------------------------------------------------------
// Orbital projection GEMM (N = 16): wave-per-16x16-tile, direct global
// fragments (Bt2 is 8 KB, L2-resident). Epilogue applies envelope.
// ---------------------------------------------------------------------------
__global__ __launch_bounds__(256) void gemm_orb2_kernel(
    const f16* __restrict__ A, const f16* __restrict__ Bt,
    const float* __restrict__ b2, const float* __restrict__ env, int envOff,
    float* __restrict__ orb, int kdet) {
  const int tid = threadIdx.x, lane = tid & 31, wv = tid >> 5;
  const bool hi = lane >= 16;
  const int l16 = lane & 15;
  const int m0 = (blockIdx.x * 8 + wv) * 16;

  v8f acc = {0.f, 0.f, 0.f, 0.f, 0.f, 0.f, 0.f, 0.f};
  const f16* arow = A + (size_t)(m0 + l16) * DMODEL;
  const f16* brow = Bt + (size_t)l16 * DMODEL + (hi ? 8 : 0);
  #pragma unroll
  for (int ks = 0; ks < 8; ++ks) {
    int k0 = ks * 32;
    union { v16h v; v8h h[2]; } fa, fb;
    fa.h[0] = *(const v8h*)(arow + k0 + (hi ? 8 : 0));
    fa.h[1] = *(const v8h*)(arow + k0 + 16 + (hi ? 8 : 0));
    fb.h[0] = *(const v8h*)(brow + k0);
    fb.h[1] = *(const v8h*)(brow + k0 + 16);
    acc = __builtin_amdgcn_wmma_f32_16x16x32_f16(
        false, fa.v, false, fb.v, (short)0, acc, false, false);
  }
  int n = l16;
  float bv = b2[n];
  #pragma unroll
  for (int r = 0; r < 8; ++r) {
    int m = m0 + r + (hi ? 8 : 0);
    int b = m >> 4, e = m & 15;
    float v = (acc[r] + bv) * env[b * NEL + envOff + e];
    orb[(((size_t)b * NDET + kdet) * 16 + e) * 16 + n] = v;
  }
}

// ---------------------------------------------------------------------------
// log|det| of 16x16 matrices, one wave per matrix, LU with partial pivoting.
// ---------------------------------------------------------------------------
__global__ __launch_bounds__(256) void slogdet16_kernel(const float* __restrict__ orb,
                                                        float* __restrict__ ld) {
  __shared__ float Mx[8][16][17];
  __shared__ int piv[8];
  __shared__ float fs[8][16];
  int tid = threadIdx.x, lane = tid & 31, wv = tid >> 5;
  int mat = blockIdx.x * 8 + wv;
  const float* src = orb + (size_t)mat * 256;
  for (int i = lane; i < 256; i += 32) Mx[wv][i >> 4][i & 15] = src[i];
  __syncthreads();
  float logsum = 0.f;
  for (int i = 0; i < 16; ++i) {
    if (lane == 0) {
      int p = i; float best = fabsf(Mx[wv][i][i]);
      for (int j = i + 1; j < 16; ++j) {
        float v = fabsf(Mx[wv][j][i]);
        if (v > best) { best = v; p = j; }
      }
      piv[wv] = p;
    }
    __syncthreads();
    int p = piv[wv];
    if (p != i && lane < 16) {
      float t = Mx[wv][i][lane];
      Mx[wv][i][lane] = Mx[wv][p][lane];
      Mx[wv][p][lane] = t;
    }
    __syncthreads();
    float d = Mx[wv][i][i];
    if (lane == 0) logsum += logf(fabsf(d) + 1e-30f);
    if (lane < 15 - i) fs[wv][lane] = Mx[wv][i + 1 + lane][i] / d;
    __syncthreads();
    int cnt = (15 - i) * 16;
    for (int t = lane; t < cnt; t += 32) {
      int j = i + 1 + (t >> 4), c = t & 15;
      Mx[wv][j][c] -= fs[wv][j - i - 1] * Mx[wv][i][c];
    }
    __syncthreads();
  }
  if (lane == 0) ld[mat] = logsum;
}

// ---------------------------------------------------------------------------
// logsumexp over dets + Jastrow, one thread per batch element.
// ---------------------------------------------------------------------------
__global__ void finalize_kernel(const float* __restrict__ ldup, const float* __restrict__ lddn,
                                const float* __restrict__ ec, const float* __restrict__ spins,
                                const float* __restrict__ apar, const float* __restrict__ aanti,
                                float* __restrict__ out) {
  int b = blockIdx.x * 256 + threadIdx.x;
  if (b >= BATCH) return;
  float t[NDET];
  float mx = -3.0e38f;
  for (int k = 0; k < NDET; ++k) {
    t[k] = ldup[b * NDET + k] + lddn[b * NDET + k];
    mx = fmaxf(mx, t[k]);
  }
  float s = 0.f;
  for (int k = 0; k < NDET; ++k) s += __expf(t[k] - mx);
  float lse = mx + logf(s);

  float ap = apar[0], aa = aanti[0];
  float lj = 0.f;
  for (int i = 0; i < NEL; ++i) {
    float xi = ec[(b * NEL + i) * 3 + 0];
    float yi = ec[(b * NEL + i) * 3 + 1];
    float zi = ec[(b * NEL + i) * 3 + 2];
    float si = spins[b * NEL + i];
    for (int j = i + 1; j < NEL; ++j) {
      float dx = xi - ec[(b * NEL + j) * 3 + 0];
      float dy = yi - ec[(b * NEL + j) * 3 + 1];
      float dz = zi - ec[(b * NEL + j) * 3 + 2];
      float r = fmaxf(sqrtf(dx * dx + dy * dy + dz * dz + 1e-20f), 1e-10f);
      bool same = si * spins[b * NEL + j] > 0.f;
      lj += same ? (-0.25f * ap * ap / (ap + r)) : (-0.5f * aa * aa / (aa + r));
    }
  }
  out[b] = lse + lj;
}

// ---------------------------------------------------------------------------
// Host orchestration.
// ---------------------------------------------------------------------------
extern "C" void kernel_launch(void* const* d_in, const int* in_sizes, int n_in,
                              void* d_out, int out_size, void* d_ws, size_t ws_size,
                              hipStream_t stream) {
  (void)in_sizes; (void)n_in; (void)out_size; (void)ws_size;
  const float* ec    = (const float*)d_in[0];
  const float* spins = (const float*)d_in[1];
  const float* atoms = (const float*)d_in[2];
  const float* Win   = (const float*)d_in[3];
  const float* bin   = (const float*)d_in[4];
  const float* Wq    = (const float*)d_in[5];
  const float* Wk    = (const float*)d_in[6];
  const float* Wv    = (const float*)d_in[7];
  const float* Wo    = (const float*)d_in[8];
  const float* Wd    = (const float*)d_in[9];
  const float* bd    = (const float*)d_in[10];
  const float* Ws1   = (const float*)d_in[11];
  const float* bs1   = (const float*)d_in[12];
  const float* Ws2   = (const float*)d_in[13];
  const float* bs2   = (const float*)d_in[14];
  const float* Wu1   = (const float*)d_in[15];
  const float* bu1   = (const float*)d_in[16];
  const float* Wu2   = (const float*)d_in[17];
  const float* bu2   = (const float*)d_in[18];
  const float* Wn1   = (const float*)d_in[19];
  const float* bn1   = (const float*)d_in[20];
  const float* Wn2   = (const float*)d_in[21];
  const float* bn2   = (const float*)d_in[22];
  const float* sigma = (const float*)d_in[23];
  const float* pienv = (const float*)d_in[24];
  const float* apar  = (const float*)d_in[25];
  const float* aanti = (const float*)d_in[26];

  char* ws = (char*)d_ws;
  size_t cur = 0;
  auto alloc = [&](size_t bytes) -> char* {
    char* p = ws + cur;
    cur = (cur + bytes + 255) & ~(size_t)255;
    return p;
  };

  f16* wQKVT = (f16*)alloc((size_t)NLAY * 768 * 256 * 2);
  f16* wOT   = (f16*)alloc((size_t)NLAY * 256 * 256 * 2);
  f16* wDT   = (f16*)alloc((size_t)NLAY * 256 * 256 * 2);
  f16* wS1T  = (f16*)alloc((size_t)NLAY * 256 * 256 * 2);
  f16* wS2T  = (f16*)alloc((size_t)NLAY * 256 * 256 * 2);
  f16* wU1T  = (f16*)alloc((size_t)NDET * 256 * 256 * 2);
  f16* wU2T  = (f16*)alloc((size_t)NDET * 16 * 256 * 2);
  f16* wN1T  = (f16*)alloc((size_t)NDET * 256 * 256 * 2);
  f16* wN2T  = (f16*)alloc((size_t)NDET * 16 * 256 * 2);
  float* h   = (float*)alloc((size_t)ROWS * DMODEL * 4);
  f16* hh    = (f16*)alloc((size_t)ROWS * DMODEL * 2);
  f16* qkv   = (f16*)alloc((size_t)ROWS * 768 * 2);
  f16* obuf  = (f16*)alloc((size_t)ROWS * DMODEL * 2);
  f16* abuf  = (f16*)alloc((size_t)ROWS * DMODEL * 2);
  float* env = (float*)alloc((size_t)ROWS * 4);
  f16* hfu   = (f16*)alloc((size_t)HROWS * DMODEL * 2);
  f16* hfd   = (f16*)alloc((size_t)HROWS * DMODEL * 2);
  float* ldup = (float*)alloc((size_t)BATCH * NDET * 4);
  float* lddn = (float*)alloc((size_t)BATCH * NDET * 4);
  // reuse large transient regions after the transformer finishes:
  float* orb_up = (float*)qkv;                                    // 32 MB
  float* orb_dn = (float*)((char*)qkv + (size_t)50331648);        // 32 MB, inside qkv region
  f16*   tdet   = abuf;                                           // 16 MB, inside abuf

  auto convL = [&](const float* src, f16* dst, int Din, int Dout) {
    int total = Din * Dout;
    conv_t_kernel<<<(total + 255) / 256, 256, 0, stream>>>(src, dst, Din, Dout);
  };

  for (int l = 0; l < NLAY; ++l) {
    size_t wo = (size_t)l * 65536;
    convL(Wq + wo, wQKVT + (size_t)l * 768 * 256 + 0 * 65536, 256, 256);
    convL(Wk + wo, wQKVT + (size_t)l * 768 * 256 + 1 * 65536, 256, 256);
    convL(Wv + wo, wQKVT + (size_t)l * 768 * 256 + 2 * 65536, 256, 256);
    convL(Wo + wo, wOT  + wo, 256, 256);
    convL(Wd + wo, wDT  + wo, 256, 256);
    convL(Ws1 + wo, wS1T + wo, 256, 256);
    convL(Ws2 + wo, wS2T + wo, 256, 256);
  }
  for (int k = 0; k < NDET; ++k) {
    convL(Wu1 + (size_t)k * 65536, wU1T + (size_t)k * 65536, 256, 256);
    convL(Wu2 + (size_t)k * 4096,  wU2T + (size_t)k * 4096,  256, 16);
    convL(Wn1 + (size_t)k * 65536, wN1T + (size_t)k * 65536, 256, 256);
    convL(Wn2 + (size_t)k * 4096,  wN2T + (size_t)k * 4096,  256, 16);
  }

  features_kernel<<<ROWS / 256, 256, 0, stream>>>(ec, spins, atoms, Win, bin,
                                                  sigma, pienv, h, env);

  for (int l = 0; l < NLAY; ++l) {
    f32_to_f16_kernel<<<4096, 256, 0, stream>>>(h, hh, (size_t)ROWS * DMODEL);
    gemm_wmma_kernel<EPI_F16><<<dim3(12, ROWS / 128), 256, 0, stream>>>(
        hh, wQKVT + (size_t)l * 768 * 256, nullptr, qkv, nullptr, ROWS, 768, 256);
    attn_kernel<<<BATCH, 256, 0, stream>>>(qkv, obuf);
    gemm_wmma_kernel<EPI_F16><<<dim3(4, ROWS / 128), 256, 0, stream>>>(
        obuf, wOT + (size_t)l * 65536, nullptr, abuf, nullptr, ROWS, 256, 256);
    gemm_wmma_kernel<EPI_TANH_ADD><<<dim3(4, ROWS / 128), 256, 0, stream>>>(
        abuf, wDT + (size_t)l * 65536, bd + l * 256, nullptr, h, ROWS, 256, 256);
    gemm_wmma_kernel<EPI_TANH_F16><<<dim3(4, ROWS / 128), 256, 0, stream>>>(
        hh, wS1T + (size_t)l * 65536, bs1 + l * 256, obuf, nullptr, ROWS, 256, 256);
    gemm_wmma_kernel<EPI_TANH_ADD><<<dim3(4, ROWS / 128), 256, 0, stream>>>(
        obuf, wS2T + (size_t)l * 65536, bs2 + l * 256, nullptr, h, ROWS, 256, 256);
  }

  split_kernel<<<(HROWS * DMODEL) / 256, 256, 0, stream>>>(h, hfu, hfd);

  for (int k = 0; k < NDET; ++k) {
    gemm_wmma_kernel<EPI_GELU_F16><<<dim3(4, HROWS / 128), 256, 0, stream>>>(
        hfu, wU1T + (size_t)k * 65536, bu1 + k * 256, tdet, nullptr, HROWS, 256, 256);
    gemm_orb2_kernel<<<HROWS / 128, 256, 0, stream>>>(
        tdet, wU2T + (size_t)k * 4096, bu2 + k * 16, env, 0, orb_up, k);
    gemm_wmma_kernel<EPI_GELU_F16><<<dim3(4, HROWS / 128), 256, 0, stream>>>(
        hfd, wN1T + (size_t)k * 65536, bn1 + k * 256, tdet, nullptr, HROWS, 256, 256);
    gemm_orb2_kernel<<<HROWS / 128, 256, 0, stream>>>(
        tdet, wN2T + (size_t)k * 4096, bn2 + k * 16, env, NUP, orb_dn, k);
  }

  slogdet16_kernel<<<(BATCH * NDET) / 8, 256, 0, stream>>>(orb_up, ldup);
  slogdet16_kernel<<<(BATCH * NDET) / 8, 256, 0, stream>>>(orb_dn, lddn);
  finalize_kernel<<<BATCH / 256, 256, 0, stream>>>(ldup, lddn, ec, spins, apar, aanti,
                                                   (float*)d_out);
}
